// HiroLRAN_73581379715573
// MI455X (gfx1250) — matrix operand order repaired and checked by reading, not compiled, not run on hardware
//
#include <hip/hip_runtime.h>
#include <hip/hip_bf16.h>

// ---------------------------------------------------------------------------
// HiroLRAN: encoder GEMMs -> linear time scan -> decoder GEMMs
// MI455X / gfx1250: wave32, v_wmma_f32_16x16x32_f16, LDS double-buffered
// staging via GLOBAL_LOAD_ASYNC_TO_LDS_B128 (ASYNCcnt tracked).
// ---------------------------------------------------------------------------

typedef __attribute__((ext_vector_type(16))) _Float16 v16h;
typedef __attribute__((ext_vector_type(8)))  _Float16 v8h;
typedef __attribute__((ext_vector_type(8)))  float    v8f;
typedef int i32x4 __attribute__((vector_size(16)));   // matches builtin pointee

#define B_DIM   32
#define T_DIM   2048
#define NTOK    (B_DIM * T_DIM)      // 65536
#define STATE   128
#define ACT     16
#define LAT     256
#define ENC     1024
#define CHUNK   8192                 // tokens per encoder/decoder chunk
#define NCHUNK  (NTOK / CHUNK)

#define LDS_STRIDE 40                // halves; 80B = 20 banks -> conflict-free

// ---- workspace layout (bytes) ---------------------------------------------
#define OFF_WTE1 ((size_t)0)                         // 1024x128  f16
#define OFF_WTE2 (OFF_WTE1 + (size_t)128*1024*2)     // 1024x1024 f16
#define OFF_WTE3 (OFF_WTE2 + (size_t)1024*1024*2)    // 256x1024  f16
#define OFF_WTD1 (OFF_WTE3 + (size_t)1024*256*2)     // 1024x256  f16
#define OFF_WTD2 (OFF_WTD1 + (size_t)256*1024*2)     // 1024x1024 f16
#define OFF_WTD3 (OFF_WTD2 + (size_t)1024*1024*2)    // 128x1024  f16
#define OFF_X16  (OFF_WTD3 + (size_t)1024*128*2)     // 65536x128 f16
#define OFF_H1   (OFF_X16  + (size_t)NTOK*STATE*2)   // 8192x1024 f16 (chunk)
#define OFF_H2   (OFF_H1   + (size_t)CHUNK*ENC*2)    // 8192x1024 f16 (chunk)
#define OFF_Z    (OFF_H2   + (size_t)CHUNK*ENC*2)    // 65536x256 f16
#define OFF_Z1   (OFF_Z    + (size_t)NTOK*LAT*2)     // 65536x256 f16
#define OFF_BU   (OFF_Z1   + (size_t)NTOK*LAT*2)     // 65536x256 f32

// ---- async global->LDS support probe --------------------------------------
#if defined(__has_builtin)
#if __has_builtin(__builtin_amdgcn_global_load_async_to_lds_b128)
#define HAVE_ASYNC_CP 1
#endif
#endif
#ifndef HAVE_ASYNC_CP
#define HAVE_ASYNC_CP 0
#endif

#if HAVE_ASYNC_CP
__device__ __forceinline__ void async_cp16(const _Float16* g, _Float16* l) {
    // (int4 AS1* src, int4 AS3* dst, imm offset, imm cpol)
    __builtin_amdgcn_global_load_async_to_lds_b128(
        (__attribute__((address_space(1))) i32x4*)(void*)(g),
        (__attribute__((address_space(3))) i32x4*)(void*)(l), 0, 0);
}
__device__ __forceinline__ void wait_async0() {
#if __has_builtin(__builtin_amdgcn_s_wait_asynccnt)
    __builtin_amdgcn_s_wait_asynccnt(0);
#else
    asm volatile("s_wait_asynccnt 0" ::: "memory");
#endif
}
#endif

__device__ __forceinline__ float fast_tanh(float x) {
    // tanh(x) = 1 - 2/(exp2(2x*log2 e)+1); saturates correctly at +-inf
    float t = __builtin_amdgcn_exp2f(x * 2.885390081777927f);
    return 1.0f - 2.0f * __builtin_amdgcn_rcpf(t + 1.0f);
}

// ---- transpose + f32->f16 convert: W(K,N) -> Wt(N,K) ----------------------
__global__ void wconv_kernel(const float* __restrict__ W,
                             _Float16* __restrict__ Wt, int K, int N) {
    int idx = blockIdx.x * 256 + threadIdx.x;
    if (idx >= K * N) return;
    int n = idx / K, k = idx - n * K;
    Wt[idx] = (_Float16)W[(size_t)k * N + n];
}

// ---- extract x (first STATE cols of padded rows) to f16, warmup rows only --
__global__ void xconv_kernel(const float* __restrict__ padded,
                             _Float16* __restrict__ x16,
                             const int* __restrict__ nwp) {
    int idx = blockIdx.x * 256 + threadIdx.x;
    int r = idx >> 7, c = idx & 127;           // NTOK x 128
    int t = r & (T_DIM - 1);
    int nw = *nwp; if (nw < 0) nw = 0;
    if (t > nw) return;                        // encoder output unused past warmup
    x16[idx] = (_Float16)padded[(size_t)r * (STATE + 2 * ACT) + c];
}

// ---- Bu = u @ B_w : (NTOK x 16) @ (16 x 256) -> f32 -----------------------
__global__ void bu_kernel(const float* __restrict__ padded,
                          const float* __restrict__ Bw,
                          float* __restrict__ bu) {
    int idx = blockIdx.x * 256 + threadIdx.x;  // NTOK*LAT threads
    int r = idx >> 8, n = idx & 255;
    const float* u = padded + (size_t)r * (STATE + 2 * ACT) + (STATE + ACT);
    float s = 0.f;
#pragma unroll
    for (int k = 0; k < ACT; ++k) s = fmaf(u[k], Bw[k * LAT + n], s);
    bu[idx] = s;
}

// ---- sequential recurrence: out[t] = a * (t<=nw ? z[t] : prev) + bu[t] ----
__global__ void scan_kernel(const _Float16* __restrict__ z,
                            const float* __restrict__ bu,
                            const float* __restrict__ a_diag,
                            const int* __restrict__ nwp,
                            _Float16* __restrict__ z1) {
    int tid = blockIdx.x * 256 + threadIdx.x;  // 8192 chains
    int b = tid >> 8, n = tid & 255;
    float a = a_diag[n];
    a = fminf(0.95f, fmaxf(-0.95f, a));
    int nw = *nwp; if (nw < 0) nw = 0;
    int nwc = nw < T_DIM - 1 ? nw : T_DIM - 1;
    size_t base = (size_t)b * T_DIM * LAT + n;
    float prev = 0.f;
    int t = 0;
    for (; t <= nwc; ++t) {                    // warmup: feed encoder state
        size_t i = base + (size_t)t * LAT;
        prev = fmaf(a, (float)z[i], bu[i]);
        z1[i] = (_Float16)prev;
    }
    for (; t < T_DIM; ++t) {                   // free-running recurrence
        size_t i = base + (size_t)t * LAT;
        prev = fmaf(a, prev, bu[i]);
        z1[i] = (_Float16)prev;
    }
}

// ---- WMMA GEMM: out = act(A(MxK) @ Wt(NxK)^T + bias) ----------------------
// 256 thr = 8 waves; block tile 128(M) x 128(N) x 32(K); LDS double-buffered.
// Wave w owns rows w*16..w*16+15 and all 128 cols: 8 f32 accumulators.
template <int KDIM, bool TANH, bool F32OUT, bool EARLY>
__global__ __launch_bounds__(256)
void gemm_wmma_kernel(const _Float16* __restrict__ A,
                      const _Float16* __restrict__ Wt,
                      const float* __restrict__ bias,
                      void* __restrict__ outp, int N,
                      int rowOff, const int* __restrict__ nwp) {
    __shared__ _Float16 sA[2][128 * LDS_STRIDE];
    __shared__ _Float16 sB[2][128 * LDS_STRIDE];

    if (EARLY) {  // skip tiles whose every token has t > nwarmup (uniform)
        int nw = *nwp; if (nw < 0) nw = 0;
        int t0 = (rowOff + (int)blockIdx.x * 128) & (T_DIM - 1);
        if (t0 > nw) return;
    }
    const int tid    = threadIdx.x;
    const int wave   = tid >> 5;
    const int lane   = tid & 31;
    const int laneHi = lane >> 4;
    const int lane16 = lane & 15;
    const int rowBase = blockIdx.x * 128;
    const int colBase = blockIdx.y * 128;

    // staging: 512 x 16B chunks each for A and B; thread handles chunk tid and tid+256
    const int r0 = tid >> 2,          q0 = tid & 3;
    const int r1 = (tid + 256) >> 2,  q1 = (tid + 256) & 3;
    const _Float16* gA0 = A  + (size_t)(rowBase + r0) * KDIM + q0 * 8;
    const _Float16* gA1 = A  + (size_t)(rowBase + r1) * KDIM + q1 * 8;
    const _Float16* gB0 = Wt + (size_t)(colBase + r0) * KDIM + q0 * 8;
    const _Float16* gB1 = Wt + (size_t)(colBase + r1) * KDIM + q1 * 8;
    const int lA0 = r0 * LDS_STRIDE + q0 * 8;
    const int lA1 = r1 * LDS_STRIDE + q1 * 8;

    v8f acc[8];
#pragma unroll
    for (int j = 0; j < 8; ++j) acc[j] = v8f{};

    constexpr int KT = KDIM / 32;

    // ---- stage k-slab 0 into buffer 0 ----
#if HAVE_ASYNC_CP
    async_cp16(gA0, &sA[0][lA0]);
    async_cp16(gA1, &sA[0][lA1]);
    async_cp16(gB0, &sB[0][lA0]);
    async_cp16(gB1, &sB[0][lA1]);
#else
    {
        v8h a0 = *(const v8h*)gA0, a1 = *(const v8h*)gA1;
        v8h b0 = *(const v8h*)gB0, b1 = *(const v8h*)gB1;
        *(v8h*)&sA[0][lA0] = a0;  *(v8h*)&sA[0][lA1] = a1;
        *(v8h*)&sB[0][lA0] = b0;  *(v8h*)&sB[0][lA1] = b1;
    }
#endif

    for (int kt = 0; kt < KT; ++kt) {
        const int buf = kt & 1;
#if HAVE_ASYNC_CP
        wait_async0();
#endif
        __syncthreads();   // stage kt visible; all waves done with buf^1

        if (kt + 1 < KT) { // stage kt+1 into other buffer, overlaps compute
            const int k = (kt + 1) * 32;
#if HAVE_ASYNC_CP
            async_cp16(gA0 + k, &sA[buf ^ 1][lA0]);
            async_cp16(gA1 + k, &sA[buf ^ 1][lA1]);
            async_cp16(gB0 + k, &sB[buf ^ 1][lA0]);
            async_cp16(gB1 + k, &sB[buf ^ 1][lA1]);
#else
            v8h a0 = *(const v8h*)(gA0 + k), a1 = *(const v8h*)(gA1 + k);
            v8h b0 = *(const v8h*)(gB0 + k), b1 = *(const v8h*)(gB1 + k);
            *(v8h*)&sA[buf ^ 1][lA0] = a0;  *(v8h*)&sA[buf ^ 1][lA1] = a1;
            *(v8h*)&sB[buf ^ 1][lA0] = b0;  *(v8h*)&sB[buf ^ 1][lA1] = b1;
#endif
        }

        // A fragment: row = wave*16+lane16; halves [0..7]=K(0..7|8..15),
        // [8..15]=K(16..23|24..31) per the 16-bit 16x32 layout.
        const _Float16* ap = &sA[buf][(wave * 16 + lane16) * LDS_STRIDE + laneHi * 8];
        v8h alo = *(const v8h*)ap;
        v8h ahi = *(const v8h*)(ap + 16);
        v16h a = __builtin_shufflevector(alo, ahi, 0, 1, 2, 3, 4, 5, 6, 7,
                                         8, 9, 10, 11, 12, 13, 14, 15);
#pragma unroll
        for (int j = 0; j < 8; ++j) {
            const _Float16* bp = &sB[buf][(j * 16 + lane16) * LDS_STRIDE + laneHi * 8];
            v8h blo = *(const v8h*)bp;
            v8h bhi = *(const v8h*)(bp + 16);
            v16h b = __builtin_shufflevector(blo, bhi, 0, 1, 2, 3, 4, 5, 6, 7,
                                             8, 9, 10, 11, 12, 13, 14, 15);
            acc[j] = __builtin_amdgcn_wmma_f32_16x16x32_f16(
                false, a, false, b, (short)0, acc[j], false, false);
        }
    }

    // C/D layout: VGPR r -> (M=r, N=lane) lanes 0-15 ; (M=8+r, N=lane-16) 16-31
#pragma unroll
    for (int j = 0; j < 8; ++j) {
        const int col = colBase + j * 16 + lane16;
        const float bj = bias[col];
#pragma unroll
        for (int r = 0; r < 8; ++r) {
            const int row = rowBase + wave * 16 + laneHi * 8 + r;
            float v = acc[j][r] + bj;
            if (TANH) v = fast_tanh(v);
            if (F32OUT)
                ((float*)outp)[(size_t)row * N + col] = v;
            else
                ((_Float16*)outp)[(size_t)row * N + col] = (_Float16)v;
        }
    }
}

// ---------------------------------------------------------------------------
extern "C" void kernel_launch(void* const* d_in, const int* in_sizes, int n_in,
                              void* d_out, int out_size, void* d_ws, size_t ws_size,
                              hipStream_t stream) {
    const float* padded = (const float*)d_in[0];
    const float* enc_w1 = (const float*)d_in[1];
    const float* enc_b1 = (const float*)d_in[2];
    const float* enc_w2 = (const float*)d_in[3];
    const float* enc_b2 = (const float*)d_in[4];
    const float* enc_w3 = (const float*)d_in[5];
    const float* enc_b3 = (const float*)d_in[6];
    const float* a_diag = (const float*)d_in[7];
    const float* B_w    = (const float*)d_in[8];
    const float* dec_w1 = (const float*)d_in[9];
    const float* dec_b1 = (const float*)d_in[10];
    const float* dec_w2 = (const float*)d_in[11];
    const float* dec_b2 = (const float*)d_in[12];
    const float* dec_w3 = (const float*)d_in[13];
    const float* dec_b3 = (const float*)d_in[14];
    const int*   nwp    = (const int*)d_in[15];
    float* out = (float*)d_out;

    char* ws = (char*)d_ws;
    _Float16* wtE1 = (_Float16*)(ws + OFF_WTE1);
    _Float16* wtE2 = (_Float16*)(ws + OFF_WTE2);
    _Float16* wtE3 = (_Float16*)(ws + OFF_WTE3);
    _Float16* wtD1 = (_Float16*)(ws + OFF_WTD1);
    _Float16* wtD2 = (_Float16*)(ws + OFF_WTD2);
    _Float16* wtD3 = (_Float16*)(ws + OFF_WTD3);
    _Float16* x16  = (_Float16*)(ws + OFF_X16);
    _Float16* h1   = (_Float16*)(ws + OFF_H1);
    _Float16* h2   = (_Float16*)(ws + OFF_H2);
    _Float16* zbuf = (_Float16*)(ws + OFF_Z);
    _Float16* z1   = (_Float16*)(ws + OFF_Z1);
    float*    bu   = (float*)(ws + OFF_BU);

    // 1) weights: transpose + f32 -> f16
    wconv_kernel<<<(128 * 1024) / 256, 256, 0, stream>>>(enc_w1, wtE1, 128, 1024);
    wconv_kernel<<<(1024 * 1024) / 256, 256, 0, stream>>>(enc_w2, wtE2, 1024, 1024);
    wconv_kernel<<<(1024 * 256) / 256, 256, 0, stream>>>(enc_w3, wtE3, 1024, 256);
    wconv_kernel<<<(256 * 1024) / 256, 256, 0, stream>>>(dec_w1, wtD1, 256, 1024);
    wconv_kernel<<<(1024 * 1024) / 256, 256, 0, stream>>>(dec_w2, wtD2, 1024, 1024);
    wconv_kernel<<<(1024 * 128) / 256, 256, 0, stream>>>(dec_w3, wtD3, 1024, 128);

    // 2) x -> f16 (warmup rows only; rest of encoder is dead code)
    xconv_kernel<<<(NTOK * STATE) / 256, 256, 0, stream>>>(padded, x16, nwp);

    // 3) encoder (chunked; tiles with t0 > nwarmup exit immediately)
    for (int c = 0; c < NCHUNK; ++c) {
        const int rowOff = c * CHUNK;
        gemm_wmma_kernel<128, true, false, true>
            <<<dim3(CHUNK / 128, 1024 / 128), 256, 0, stream>>>(
                x16 + (size_t)rowOff * STATE, wtE1, enc_b1, h1, 1024, rowOff, nwp);
        gemm_wmma_kernel<1024, true, false, true>
            <<<dim3(CHUNK / 128, 1024 / 128), 256, 0, stream>>>(
                h1, wtE2, enc_b2, h2, 1024, rowOff, nwp);
        gemm_wmma_kernel<1024, true, false, true>
            <<<dim3(CHUNK / 128, 256 / 128), 256, 0, stream>>>(
                h2, wtE3, enc_b3, zbuf + (size_t)rowOff * LAT, 256, rowOff, nwp);
    }

    // 4) Bu for all tokens, then the sequential scan (8192 parallel chains)
    bu_kernel<<<(NTOK * LAT) / 256, 256, 0, stream>>>(padded, B_w, bu);
    scan_kernel<<<(B_DIM * LAT) / 256, 256, 0, stream>>>(zbuf, bu, a_diag, nwp, z1);

    // 5) decoder (all tokens)
    for (int c = 0; c < NCHUNK; ++c) {
        const size_t rowOff = (size_t)c * CHUNK;
        gemm_wmma_kernel<256, true, false, false>
            <<<dim3(CHUNK / 128, 1024 / 128), 256, 0, stream>>>(
                z1 + rowOff * LAT, wtD1, dec_b1, h1, 1024, 0, nullptr);
        gemm_wmma_kernel<1024, true, false, false>
            <<<dim3(CHUNK / 128, 1024 / 128), 256, 0, stream>>>(
                h1, wtD2, dec_b2, h2, 1024, 0, nullptr);
        gemm_wmma_kernel<1024, false, true, false>
            <<<dim3(CHUNK / 128, 128 / 128), 256, 0, stream>>>(
                h2, wtD3, dec_b3, out + rowOff * STATE, 128, 0, nullptr);
    }
}
